// ECGLSTM_7052336300072
// MI455X (gfx1250) — matrix operand order repaired and checked by reading, not compiled, not run on hardware
//
#include <hip/hip_runtime.h>
#include <hip/hip_bf16.h>

// Problem constants (from the reference setup_inputs)
#define HID     64
#define BATCH   256
#define TSTEPS  5000
#define NOUT    128
#define ROWS    16          // batch rows per workgroup
#define CT      64          // x timestep chunk held in LDS
#define HSTRIDE 72          // padded LDS row stride for h (f16 elems) = 144B, bank-spread

typedef __attribute__((ext_vector_type(16))) _Float16 v16h;
typedef __attribute__((ext_vector_type(8)))  float    v8f;

union V16H {
    v16h     v;
    _Float16 h[16];
    uint4    q[2];
};

// ---- Branch-free, native-instruction activations -------------------------
// gfx1250 has v_tanh_f32; fall back to v_exp_f32 (exp2) + v_rcp_f32 forms.
__device__ __forceinline__ float fast_exp2(float x) {
#if __has_builtin(__builtin_amdgcn_exp2f)
    return __builtin_amdgcn_exp2f(x);
#else
    return exp2f(x);
#endif
}
__device__ __forceinline__ float fast_rcp(float x) {
#if __has_builtin(__builtin_amdgcn_rcpf)
    return __builtin_amdgcn_rcpf(x);
#else
    return 1.0f / x;
#endif
}
__device__ __forceinline__ float fast_tanh(float x) {
#if __has_builtin(__builtin_amdgcn_tanhf)
    return __builtin_amdgcn_tanhf(x);
#else
    const float ax = __builtin_fabsf(x);
    const float t  = fast_exp2(-2.8853900817779268f * ax);   // exp(-2|x|)
    const float r  = 1.0f - 2.0f * t * fast_rcp(1.0f + t);   // tanh(|x|)
    return __builtin_copysignf(r, x);
#endif
}
__device__ __forceinline__ float fast_sigmoid(float x) {
#if __has_builtin(__builtin_amdgcn_tanhf)
    return __builtin_fmaf(0.5f, __builtin_amdgcn_tanhf(0.5f * x), 0.5f);
#else
    // exp2 overflow -> inf -> rcp -> 0 : saturates correctly, branch-free
    return fast_rcp(1.0f + fast_exp2(-1.4426950408889634f * x));
#endif
}

__global__ __launch_bounds__(128)
void ECGLSTM_kernel(const float* __restrict__ x,      // [BATCH, TSTEPS]
                    const float* __restrict__ W_ih,   // [4H, 1]
                    const float* __restrict__ W_hh,   // [4H, H]
                    const float* __restrict__ b_ih,   // [4H]
                    const float* __restrict__ b_hh,   // [4H]
                    const float* __restrict__ W_fc,   // [128, 64]
                    const float* __restrict__ b_fc,   // [128]
                    float* __restrict__ out)          // [BATCH, 128]
{
    __shared__ alignas(16) _Float16 h_lds[2][ROWS * HSTRIDE]; // double-buffered h tile
    __shared__ alignas(16) float    x_lds[ROWS][CT];          // x chunk
    __shared__ alignas(16) float    hf_lds[ROWS][HID];        // h_last f32 (FC epilogue)

    const int tid  = threadIdx.x;
    const int wave = tid >> 5;       // 0..3
    const int lane = tid & 31;
    const int ln   = lane & 15;      // N within tile / row M for A
    const int half = lane >> 4;      // 0 or 1
    const int base = blockIdx.x * ROWS;

    // ---- Loop-invariant W_hh^T fragments (B-matrix, f16) into registers ----
    // Wave w owns gate columns n = g*64 + 16w + ln for g = i,f,g,o.
    // 16-bit B 32x16 layout: lane L holds N=L%16; lanes 0-15 -> K 0..7,16..23,
    // lanes 16-31 -> K 8..15,24..31 of the 32-wide K tile; 2 K per VGPR.
    V16H  bfrag[4][2];
    float wih_g[4], bs_g[4];
#pragma unroll
    for (int g = 0; g < 4; ++g) {
        const int    n    = g * HID + wave * 16 + ln;
        const float* wrow = W_hh + n * HID;
        wih_g[g] = W_ih[n];
        bs_g[g]  = b_ih[n] + b_hh[n];
#pragma unroll
        for (int kh = 0; kh < 2; ++kh) {
            const int kb = kh * 32 + half * 8;
#pragma unroll
            for (int e = 0; e < 16; ++e)
                bfrag[g][kh].h[e] = (_Float16)wrow[kb + e + (e & 8)];
        }
    }

    // zero h_{-1} (read buffer 0 of step 0)
    for (int i = tid; i < ROWS * HSTRIDE; i += 128) h_lds[0][i] = (_Float16)0.0f;
    __syncthreads();

    v8f cstate = {};   // c tile slice in C/D layout (M = r + 8*half, N = 16w+ln)
    v8f hstate = {};
    int pb = 0;        // ping-pong read-buffer index

    for (int t = 0; t < TSTEPS; ++t) {
        // ---- refill x chunk every CT steps ----
        if ((t & (CT - 1)) == 0) {
            for (int idx = tid; idx < ROWS * CT; idx += 128) {
                const int i  = idx >> 6;          // /CT
                const int j  = idx & (CT - 1);
                const int tt = t + j;
                x_lds[i][j] = (tt < TSTEPS)
                    ? x[(size_t)(base + i) * TSTEPS + tt] : 0.0f;
            }
            __syncthreads();
        }

        // ---- A fragments of h_{t-1} from LDS (16-bit A 16x32 layout) ----
        // lane L: row M = L%16; contiguous K runs of 8 at half*8 and half*8+16.
        V16H a0, a1;
        {
            const _Float16* hrow = &h_lds[pb][ln * HSTRIDE];
            const int k0 = half * 8;
            a0.q[0] = *(const uint4*)&hrow[k0];
            a0.q[1] = *(const uint4*)&hrow[k0 + 16];
            a1.q[0] = *(const uint4*)&hrow[32 + k0];
            a1.q[1] = *(const uint4*)&hrow[32 + k0 + 16];
        }

        // per-row x values for this step (rows m = r + 8*half)
        const int tj = t & (CT - 1);
        float xv[8];
#pragma unroll
        for (int r = 0; r < 8; ++r) xv[r] = x_lds[r + 8 * half][tj];

        // ---- gates = (x*W_ih + b) + h @ W_hh^T  via 8 WMMAs ----
        v8f acc[4];
#pragma unroll
        for (int g = 0; g < 4; ++g) {
            v8f a;
#pragma unroll
            for (int r = 0; r < 8; ++r) a[r] = fmaf(xv[r], wih_g[g], bs_g[g]);
            a = __builtin_amdgcn_wmma_f32_16x16x32_f16(
                    false, a0.v, false, bfrag[g][0].v, (short)0, a, false, false);
            a = __builtin_amdgcn_wmma_f32_16x16x32_f16(
                    false, a1.v, false, bfrag[g][1].v, (short)0, a, false, false);
            acc[g] = a;
        }

        // ---- elementwise LSTM cell (wave-local); write h_t to other buffer ----
        _Float16* hdst = &h_lds[pb ^ 1][0];
        const int col  = wave * 16 + ln;
#pragma unroll
        for (int r = 0; r < 8; ++r) {
            const float ig = fast_sigmoid(acc[0][r]);
            const float fg = fast_sigmoid(acc[1][r]);
            const float gg = fast_tanh(acc[2][r]);
            const float og = fast_sigmoid(acc[3][r]);
            const float cc = fg * cstate[r] + ig * gg;
            cstate[r] = cc;
            const float hh = og * fast_tanh(cc);
            hstate[r] = hh;
            const int m = r + 8 * half;
            hdst[m * HSTRIDE + col] = (_Float16)hh;
        }
        __syncthreads();   // h_t stores visible; old reads done before next overwrite
        pb ^= 1;
    }

    // ---- publish f32 h_last tile, then FC epilogue: relu(h @ W_fc^T + b) ----
#pragma unroll
    for (int r = 0; r < 8; ++r) {
        const int m = r + 8 * half;
        hf_lds[m][wave * 16 + ln] = hstate[r];
    }
    __syncthreads();

    {
        const int    n  = tid;              // 0..127 output features
        const float  bn = b_fc[n];
        const float* wf = W_fc + n * HID;
        for (int m = 0; m < ROWS; ++m) {
            float a2 = bn;
#pragma unroll
            for (int k = 0; k < HID; ++k) a2 = fmaf(hf_lds[m][k], wf[k], a2);
            out[(size_t)(base + m) * NOUT + n] = fmaxf(a2, 0.0f);
        }
    }
}

extern "C" void kernel_launch(void* const* d_in, const int* in_sizes, int n_in,
                              void* d_out, int out_size, void* d_ws, size_t ws_size,
                              hipStream_t stream) {
    (void)in_sizes; (void)n_in; (void)out_size; (void)d_ws; (void)ws_size;
    const float* x    = (const float*)d_in[0];
    const float* W_ih = (const float*)d_in[1];
    const float* W_hh = (const float*)d_in[2];
    const float* b_ih = (const float*)d_in[3];
    const float* b_hh = (const float*)d_in[4];
    const float* W_fc = (const float*)d_in[5];
    const float* b_fc = (const float*)d_in[6];
    float* out = (float*)d_out;

    dim3 grid(BATCH / ROWS);   // 16 workgroups, one per 16-row batch tile
    dim3 block(128);           // 4 wave32s
    ECGLSTM_kernel<<<grid, block, 0, stream>>>(x, W_ih, W_hh, b_ih, b_hh,
                                               W_fc, b_fc, out);
}
